// RecurrentEncoder_52587579572263
// MI455X (gfx1250) — compile-verified, matrix operand order checked by reading, not run on hardware
//
#include <hip/hip_runtime.h>
#include <hip/hip_bf16.h>
#include <math.h>
#include <stdint.h>

// ---------------------------------------------------------------------------
// RecurrentEncoder for MI455X (gfx1250, wave32, WMMA).
// 512-step scan; per step:
//   z = r @ Wr   bf16 WMMA 16x16x32, f32 acc, split-K x4, 2 N-tiles/wave,
//                async double-buffered B staging into LDS when available
//   s = topk_scatter(z, k=819)   radix-select per row, fuses 4-way K-partial sum
//   u = tanh(x_t @ Wi + s)       bf16 WMMA, 2 N-tiles/wave, even/odd K chains
//   r = u / (||u|| + 1e-6)       emits f32 carry (d_out) + bf16 carry
// Weights packed ONCE per launch into bf16 B-operand lane layout.
// ---------------------------------------------------------------------------

typedef __attribute__((ext_vector_type(16))) __bf16 v16bf;
typedef __attribute__((ext_vector_type(8)))  float  v8f;
typedef int v4i __attribute__((vector_size(16)));   // matches async builtin param

#define AS1 __attribute__((address_space(1)))
#define AS3 __attribute__((address_space(3)))

#if defined(__gfx1250__) && __has_builtin(__builtin_amdgcn_global_load_async_to_lds_b128) && __has_builtin(__builtin_amdgcn_s_wait_asynccnt)
#define USE_ASYNC_LDS 1
#else
#define USE_ASYNC_LDS 0
#endif

#define KSPLIT 4          // split-K factor for the recurrent GEMM
#define SEG_STRIDE (64 * 2048)

union BF16x16 {
    v16bf v;
    uint4 q[2];
    unsigned short u[16];
};

__device__ __forceinline__ unsigned short f2bf(float f) {
    unsigned u = __float_as_uint(f);
    unsigned r = u + 0x7FFFu + ((u >> 16) & 1u);   // round-to-nearest-even
    return (unsigned short)(r >> 16);
}

// ---------------------------------------------------------------------------
// Pack row-major f32 weight [K][N] into bf16 B-operand lane layout:
// Bp[((kt*ntilesN + tn)*32 + lane)*16 ..], lane<16: col=tn*16+lane,
// K = kt*32 + {0..7,16..23}; lane>=16: col, K = kt*32 + {8..15,24..31}.
// ---------------------------------------------------------------------------
__global__ __launch_bounds__(256)
void pack_b_kernel(const float* __restrict__ W, unsigned short* __restrict__ Bp,
                   int K, int N) {
    int ntilesN = N >> 4;
    int ntilesK = K >> 5;
    int total   = ntilesK * ntilesN * 32;
    int idx = blockIdx.x * 256 + threadIdx.x;
    if (idx >= total) return;
    int lane    = idx & 31;
    int tileIdx = idx >> 5;
    int tn = tileIdx % ntilesN;
    int kt = tileIdx / ntilesN;
    int col    = tn * 16 + (lane & 15);
    int kShift = (lane < 16) ? 0 : 8;
    unsigned short* out = Bp + (size_t)idx * 16;
#pragma unroll
    for (int j = 0; j < 8; ++j) {
        int k0 = kt * 32 + kShift + j;
        int k1 = k0 + 16;
        out[j]     = f2bf(W[(size_t)k0 * N + col]);
        out[8 + j] = f2bf(W[(size_t)k1 * N + col]);
    }
}

__global__ __launch_bounds__(256)
void zero_u16_kernel(unsigned short* __restrict__ p, int n) {
    int i = blockIdx.x * 256 + threadIdx.x;
    if (i < n) p[i] = 0;
}

// ---------------------------------------------------------------------------
// GEMM1: zpart[kseg] = r_bf16 @ Wr over one K segment.
// grid (16, 4, KSPLIT), block 128 = 4 waves; each wave: one 16-row stripe x
// two 16-col tiles, K-chain of 16 WMMAs per tile (2 independent acc chains).
// B operand staged through LDS with async copies (double buffered) when the
// toolchain exposes the gfx1250 async builtins; direct b128 loads otherwise.
// ---------------------------------------------------------------------------
__global__ __launch_bounds__(128)
void gemm_rec_kernel(const unsigned short* __restrict__ A,
                     const unsigned short* __restrict__ Bp,
                     float* __restrict__ zpart, int N, int K) {
    const int lane = threadIdx.x & 31;
    const int wave = threadIdx.x >> 5;
    const int tn0  = (blockIdx.x * 4 + wave) * 2;   // first of 2 N-tiles
    const int tm   = blockIdx.y;
    const int kseg = blockIdx.z;
    const int ntilesN = N >> 4;
    const int segKt = (K >> 5) / KSPLIT;            // 16
    const int ktBeg = kseg * segKt;
    const int ktEnd = ktBeg + segKt;
    const int row    = tm * 16 + (lane & 15);
    const int kShift = (lane < 16) ? 0 : 8;

    v8f acc0 = {}, acc1 = {};

#if USE_ASYNC_LDS
    // [wave][buf][2 tiles * 1024B]; each lane owns 32B per tile.
    __shared__ __align__(16) unsigned char ldsB[4][2][2048];

    auto stageB = [&](int kt, int buf) {
        const unsigned short* g = Bp + (((size_t)kt * ntilesN + tn0) * 32 + lane) * 16;
        unsigned char* l = &ldsB[wave][buf][lane * 32];
        AS1 v4i* gp0 = (AS1 v4i*)(uintptr_t)(const void*)g;
        AS1 v4i* gp1 = (AS1 v4i*)(uintptr_t)(const void*)(g + 512);
        AS3 v4i* lp0 = (AS3 v4i*)(unsigned)(uintptr_t)(void*)l;
        AS3 v4i* lp1 = (AS3 v4i*)(unsigned)(uintptr_t)(void*)(l + 1024);
        __builtin_amdgcn_global_load_async_to_lds_b128(gp0, lp0, 0, 0);
        __builtin_amdgcn_global_load_async_to_lds_b128(gp0, lp0, 16, 0);
        __builtin_amdgcn_global_load_async_to_lds_b128(gp1, lp1, 0, 0);
        __builtin_amdgcn_global_load_async_to_lds_b128(gp1, lp1, 16, 0);
    };

    stageB(ktBeg, 0);
    int buf = 0;
    for (int kt = ktBeg; kt < ktEnd; ++kt) {
        if (kt + 1 < ktEnd) {
            stageB(kt + 1, buf ^ 1);
            __builtin_amdgcn_s_wait_asynccnt(4);   // current buf's 4 copies done
        } else {
            __builtin_amdgcn_s_wait_asynccnt(0);
        }
        const uint4* ap = (const uint4*)(A + (size_t)row * K + kt * 32 + kShift);
        BF16x16 a, b0, b1;
        a.q[0] = ap[0];                            // K = kShift + 0..7
        a.q[1] = ap[2];                            // K = kShift + 16..23
        const uint4* lb0 = (const uint4*)&ldsB[wave][buf][lane * 32];
        const uint4* lb1 = (const uint4*)&ldsB[wave][buf][1024 + lane * 32];
        b0.q[0] = lb0[0]; b0.q[1] = lb0[1];
        b1.q[0] = lb1[0]; b1.q[1] = lb1[1];
        acc0 = __builtin_amdgcn_wmma_f32_16x16x32_bf16(
            false, a.v, false, b0.v, (short)0, acc0, false, false);
        acc1 = __builtin_amdgcn_wmma_f32_16x16x32_bf16(
            false, a.v, false, b1.v, (short)0, acc1, false, false);
        buf ^= 1;
    }
#else
    for (int kt = ktBeg; kt < ktEnd; ++kt) {
        const uint4* ap = (const uint4*)(A + (size_t)row * K + kt * 32 + kShift);
        const unsigned short* g = Bp + (((size_t)kt * ntilesN + tn0) * 32 + lane) * 16;
        if (kt + 1 < ktEnd)
            __builtin_prefetch((const void*)(g + (size_t)ntilesN * 512), 0, 1);
        BF16x16 a, b0, b1;
        a.q[0] = ap[0];
        a.q[1] = ap[2];
        const uint4* bp = (const uint4*)g;
        b0.q[0] = bp[0]; b0.q[1] = bp[1];
        b1.q[0] = bp[32]; b1.q[1] = bp[33];        // +256 ushorts = packed slot of lane+16
        acc0 = __builtin_amdgcn_wmma_f32_16x16x32_bf16(
            false, a.v, false, b0.v, (short)0, acc0, false, false);
        acc1 = __builtin_amdgcn_wmma_f32_16x16x32_bf16(
            false, a.v, false, b1.v, (short)0, acc1, false, false);
    }
#endif

    float* zp = zpart + (size_t)kseg * SEG_STRIDE;
    int col0  = tn0 * 16 + (lane & 15);
    int rbase = tm * 16 + ((lane < 16) ? 0 : 8);
#pragma unroll
    for (int i = 0; i < 8; ++i) {
        zp[(size_t)(rbase + i) * N + col0]      = acc0[i];
        zp[(size_t)(rbase + i) * N + col0 + 16] = acc1[i];
    }
}

// ---------------------------------------------------------------------------
// Top-k scatter (fuses split-K reduction): per row sum the KSPLIT partials,
// cache values + monotone-mapped keys in LDS, 4-pass MSB radix select for the
// exact k-th threshold, keep >thr plus first (k - count_gt) ties, else 0.
// ---------------------------------------------------------------------------
__global__ __launch_bounds__(256)
void topk_kernel(const float* __restrict__ zpart, float* __restrict__ s,
                 int rec, int k) {
    int row = blockIdx.x;
    const float* z0 = zpart + (size_t)row * rec;
    float*       sr = s     + (size_t)row * rec;

    __shared__ float    vals[2048];
    __shared__ unsigned keys[2048];
    __shared__ int hist[256];
    __shared__ unsigned sh_prefix;
    __shared__ int sh_remaining;
    __shared__ int sh_tie;

    int tid = threadIdx.x;
    for (int i = tid; i < rec; i += 256) {
        float v = z0[i];
#pragma unroll
        for (int g = 1; g < KSPLIT; ++g)
            v += z0[(size_t)g * SEG_STRIDE + i];
        vals[i] = v;
        unsigned u = __float_as_uint(v);
        keys[i] = u ^ ((u & 0x80000000u) ? 0xFFFFFFFFu : 0x80000000u);
    }
    if (tid == 0) { sh_prefix = 0u; sh_remaining = k; sh_tie = 0; }
    __syncthreads();

    for (int pass = 3; pass >= 0; --pass) {
        if (tid < 256) hist[tid] = 0;
        __syncthreads();
        unsigned prefix = sh_prefix;
        unsigned hiMask = (pass == 3) ? 0u : (0xFFFFFFFFu << ((pass + 1) * 8));
        for (int i = tid; i < rec; i += 256) {
            unsigned key = keys[i];
            if ((key & hiMask) == prefix)
                atomicAdd(&hist[(key >> (pass * 8)) & 0xFF], 1);
        }
        __syncthreads();
        if (tid == 0) {
            int rem = sh_remaining;
            int d = 255;
            for (; d > 0; --d) {
                if (hist[d] >= rem) break;
                rem -= hist[d];
            }
            sh_remaining = rem;
            sh_prefix = prefix | ((unsigned)d << (pass * 8));
        }
        __syncthreads();
    }

    unsigned thr = sh_prefix;
    int ties = sh_remaining;
    for (int i = tid; i < rec; i += 256) {
        unsigned key = keys[i];
        float v = 0.0f;
        if (key > thr)       v = vals[i];
        else if (key == thr) { if (atomicAdd(&sh_tie, 1) < ties) v = vals[i]; }
        sr[i] = v;
    }
}

// ---------------------------------------------------------------------------
// GEMM2 + epilogue: u = tanh(x_t @ Wi + s). A is f32 (x slice), converted to
// bf16 in-register; each wave: 2 N-tiles x even/odd K chains = 4 independent
// WMMA accumulator chains; A conversion shared across the 2 N-tiles.
// ---------------------------------------------------------------------------
__global__ __launch_bounds__(128)
void gemm_in_tanh_kernel(const float* __restrict__ x, int t,
                         const unsigned short* __restrict__ Bp,
                         const float* __restrict__ s, float* __restrict__ u,
                         int N, int K, int R) {
    const int lane = threadIdx.x & 31;
    const int wave = threadIdx.x >> 5;
    const int tn0  = (blockIdx.x * 4 + wave) * 2;
    const int tm   = blockIdx.y;
    const int ntilesN = N >> 4;
    const int brow   = tm * 16 + (lane & 15);
    const int kShift = (lane < 16) ? 0 : 8;
    const float* xrow = x + ((size_t)brow * R + t) * K;

    v8f e0 = {}, o0 = {}, e1 = {}, o1 = {};
    const int nkt = K >> 5;                       // 16
    for (int kt = 0; kt < nkt; kt += 2) {
#pragma unroll
        for (int h = 0; h < 2; ++h) {             // h=0: even chain, h=1: odd
            int kb = (kt + h) * 32 + kShift;
            float4 f0 = *(const float4*)(xrow + kb);
            float4 f1 = *(const float4*)(xrow + kb + 4);
            float4 f2 = *(const float4*)(xrow + kb + 16);
            float4 f3 = *(const float4*)(xrow + kb + 20);
            BF16x16 a, b0, b1;
            a.u[0]  = f2bf(f0.x); a.u[1]  = f2bf(f0.y); a.u[2]  = f2bf(f0.z); a.u[3]  = f2bf(f0.w);
            a.u[4]  = f2bf(f1.x); a.u[5]  = f2bf(f1.y); a.u[6]  = f2bf(f1.z); a.u[7]  = f2bf(f1.w);
            a.u[8]  = f2bf(f2.x); a.u[9]  = f2bf(f2.y); a.u[10] = f2bf(f2.z); a.u[11] = f2bf(f2.w);
            a.u[12] = f2bf(f3.x); a.u[13] = f2bf(f3.y); a.u[14] = f2bf(f3.z); a.u[15] = f2bf(f3.w);
            const uint4* bp = (const uint4*)(Bp + (((size_t)(kt + h) * ntilesN + tn0) * 32 + lane) * 16);
            b0.q[0] = bp[0];  b0.q[1] = bp[1];
            b1.q[0] = bp[32]; b1.q[1] = bp[33];
            if (h == 0) {
                e0 = __builtin_amdgcn_wmma_f32_16x16x32_bf16(
                    false, a.v, false, b0.v, (short)0, e0, false, false);
                e1 = __builtin_amdgcn_wmma_f32_16x16x32_bf16(
                    false, a.v, false, b1.v, (short)0, e1, false, false);
            } else {
                o0 = __builtin_amdgcn_wmma_f32_16x16x32_bf16(
                    false, a.v, false, b0.v, (short)0, o0, false, false);
                o1 = __builtin_amdgcn_wmma_f32_16x16x32_bf16(
                    false, a.v, false, b1.v, (short)0, o1, false, false);
            }
        }
    }

    int col0  = tn0 * 16 + (lane & 15);
    int rbase = tm * 16 + ((lane < 16) ? 0 : 8);
#pragma unroll
    for (int i = 0; i < 8; ++i) {
        size_t off0 = (size_t)(rbase + i) * N + col0;
        size_t off1 = off0 + 16;
        u[off0] = tanhf(e0[i] + o0[i] + s[off0]);
        u[off1] = tanhf(e1[i] + o1[i] + s[off1]);
    }
}

// ---------------------------------------------------------------------------
// Row L2-normalize: r = u / (||u|| + 1e-6); writes f32 carry (d_out) + bf16.
// ---------------------------------------------------------------------------
__global__ __launch_bounds__(256)
void normalize_kernel(const float* __restrict__ u, float* __restrict__ r_out,
                      unsigned short* __restrict__ r_bf, int N) {
    int row = blockIdx.x;
    __shared__ float red[256];
    float sum = 0.0f;
    for (int i = threadIdx.x; i < N; i += 256) {
        float v = u[(size_t)row * N + i];
        sum += v * v;
    }
    red[threadIdx.x] = sum;
    __syncthreads();
    for (int off = 128; off > 0; off >>= 1) {
        if (threadIdx.x < off) red[threadIdx.x] += red[threadIdx.x + off];
        __syncthreads();
    }
    float scale = 1.0f / (sqrtf(red[0]) + 1e-6f);
    for (int i = threadIdx.x; i < N; i += 256) {
        size_t off = (size_t)row * N + i;
        float v = u[off] * scale;
        r_out[off] = v;
        r_bf[off]  = f2bf(v);
    }
}

// ---------------------------------------------------------------------------
// Launcher. d_in[0]=x f32[64][512][512], d_in[1]=W_input f32[512][2048],
// d_in[2]=W_recurrent f32[2048][2048]. d_out = r_final f32[64][2048].
// ---------------------------------------------------------------------------
extern "C" void kernel_launch(void* const* d_in, const int* in_sizes, int n_in,
                              void* d_out, int out_size, void* d_ws, size_t ws_size,
                              hipStream_t stream) {
    (void)in_sizes; (void)n_in; (void)out_size; (void)ws_size;
    const int Bsz = 64, E = 512, REC = 2048, R = 512;
    const int K_TOP = 819; // int(2048 * 0.4)

    const float* x  = (const float*)d_in[0];
    const float* Wi = (const float*)d_in[1];
    const float* Wr = (const float*)d_in[2];
    float* r_out = (float*)d_out;

    char* ws = (char*)d_ws;
    unsigned short* BpR   = (unsigned short*)(ws);                  //  8 MB packed Wr
    unsigned short* BpI   = (unsigned short*)(ws + 8388608);        //  2 MB packed Wi
    float* zpart          = (float*)(ws + 10485760);                //  2 MB (4 partials)
    float* s              = (float*)(ws + 12582912);                // 512 KB
    float* u              = (float*)(ws + 13107200);                // 512 KB
    unsigned short* rbf   = (unsigned short*)(ws + 13631488);       // 256 KB

    // One-time per launch: pack weights to bf16 B-operand layout, zero carry.
    {
        int totalR = (REC / 32) * (REC / 16) * 32;   // 262144
        pack_b_kernel<<<(totalR + 255) / 256, 256, 0, stream>>>(Wr, BpR, REC, REC);
        int totalI = (E / 32) * (REC / 16) * 32;     // 65536
        pack_b_kernel<<<(totalI + 255) / 256, 256, 0, stream>>>(Wi, BpI, E, REC);
        zero_u16_kernel<<<(Bsz * REC + 255) / 256, 256, 0, stream>>>(rbf, Bsz * REC);
    }

    dim3 grid_rec(16, Bsz / 16, KSPLIT);   // 64 tile-pairs x 4 M-tiles x 4 K-segs
    dim3 grid_in(16, Bsz / 16);            // 64 tile-pairs x 4 M-tiles
    for (int t = 0; t < R; ++t) {
        gemm_rec_kernel<<<grid_rec, 128, 0, stream>>>(rbf, BpR, zpart, REC, REC);
        topk_kernel<<<Bsz, 256, 0, stream>>>(zpart, s, REC, K_TOP);
        gemm_in_tanh_kernel<<<grid_in, 128, 0, stream>>>(x, t, BpI, s, u, REC, E, R);
        normalize_kernel<<<Bsz, 256, 0, stream>>>(u, r_out, rbf, REC);
    }
}
